// TransformerBlock_62225486184814
// MI455X (gfx1250) — matrix operand order repaired and checked by reading, not compiled, not run on hardware
//
#include <hip/hip_runtime.h>
#include <math.h>

// ---- problem constants (match reference) ----
#define Dm    1024
#define Hn    16
#define HDm   64
#define Sm    2048
#define Bb    4
#define ROWS  (Bb * Sm)       // 8192
#define DFF   (4 * Dm)        // 4096

typedef __attribute__((ext_vector_type(16))) __bf16 v16bf;
typedef __attribute__((ext_vector_type(8)))  float  v8f;
typedef unsigned int u32;
// exact parameter type expected by the async-LDS builtin (from hipcc diagnostic)
typedef __attribute__((__vector_size__(4 * sizeof(int)))) int v4i_t;

union Frag { v16bf v; u32 u[8]; };

// ---- gfx1250 async global->LDS path (guarded; falls back to manual copies) ----
#if defined(__has_builtin)
#  if __has_builtin(__builtin_amdgcn_global_load_async_to_lds_b128) && \
      __has_builtin(__builtin_amdgcn_s_wait_asynccnt)
#    define USE_ASYNC_LDS 1
#  endif
#endif
#ifndef USE_ASYNC_LDS
#  define USE_ASYNC_LDS 0
#endif

// copy one 16-byte chunk (8 bf16) from global to LDS
__device__ __forceinline__ void copy16(const __bf16* g, __bf16* l) {
#if USE_ASYNC_LDS
    __bf16* gn = const_cast<__bf16*>(g);
    __builtin_amdgcn_global_load_async_to_lds_b128((v4i_t*)gn, (v4i_t*)l, 0, 0);
#else
    const u32* s = (const u32*)g;
    u32* d = (u32*)l;
    d[0] = s[0]; d[1] = s[1]; d[2] = s[2]; d[3] = s[3];
#endif
}

// 16-bit A-matrix 16x32 — contiguous K order matching B: lane half selects K 16..31; VGPR p holds a K pair.
__device__ __forceinline__ int afrag_base(int p, int half) {
    return half * 16 + 2 * p;
}

__device__ __forceinline__ float gelu_tanh_f(float x) {
    const float c = 0.7978845608028654f; // sqrt(2/pi)
    float t = tanhf(c * (x + 0.044715f * x * x * x));
    return 0.5f * x * (1.0f + t);
}

// ---------------- fused cast fp32->bf16 + transpose (for weights) ----------------
// in: [R][C] fp32 row-major   out: [C][R] bf16 row-major
__global__ void cast_transpose_bf16_kernel(const float* __restrict__ in,
                                           __bf16* __restrict__ out, int R, int C) {
    __shared__ float tile[32][33];
    const int bx = blockIdx.x * 32;   // along C
    const int by = blockIdx.y * 32;   // along R
    const int tx = threadIdx.x;       // 0..31
    const int ty = threadIdx.y;       // 0..7
#pragma unroll
    for (int j = 0; j < 4; ++j)
        tile[ty + j * 8][tx] = in[(size_t)(by + ty + j * 8) * C + bx + tx];
    __syncthreads();
#pragma unroll
    for (int j = 0; j < 4; ++j)
        out[(size_t)(bx + ty + j * 8) * R + by + tx] = (__bf16)tile[tx][ty + j * 8];
}

// ---------------- LayerNorm (fp32 in, bf16 out), one row per block ----------------
__global__ void ln_bf16_kernel(const float* __restrict__ x, const float* __restrict__ g,
                               const float* __restrict__ be, __bf16* __restrict__ out) {
    const int row = blockIdx.x;
    const float* xr = x + (size_t)row * Dm;
    float v[4], s = 0.f, s2 = 0.f;
#pragma unroll
    for (int i = 0; i < 4; ++i) {
        v[i] = xr[threadIdx.x + i * 256];
        s += v[i]; s2 += v[i] * v[i];
    }
#pragma unroll
    for (int off = 16; off > 0; off >>= 1) {
        s  += __shfl_xor(s,  off, 32);
        s2 += __shfl_xor(s2, off, 32);
    }
    __shared__ float as_[8], as2_[8];
    const int wid = threadIdx.x >> 5, lane = threadIdx.x & 31;
    if (lane == 0) { as_[wid] = s; as2_[wid] = s2; }
    __syncthreads();
    float S = 0.f, S2 = 0.f;
#pragma unroll
    for (int j = 0; j < 8; ++j) { S += as_[j]; S2 += as2_[j]; }
    const float mean = S * (1.0f / Dm);
    const float var  = S2 * (1.0f / Dm) - mean * mean;
    const float rstd = rsqrtf(var + 1e-5f);
#pragma unroll
    for (int i = 0; i < 4; ++i) {
        const int c = threadIdx.x + i * 256;
        out[(size_t)row * Dm + c] = (__bf16)((v[i] - mean) * rstd * g[c] + be[c]);
    }
}

// ---------------- bf16 WMMA GEMM, async double-buffered, fused epilogue ----------
// C[M,N] = A[M,K](bf16,row-major) @ Bt[N,K](bf16, pre-transposed) (+bias)(+gelu)(+resid)
#define BM 128
#define BN 128
#define BK 32
#define PAD 8   // 16B: keeps async b128 LDS dests aligned, staggers banks

__launch_bounds__(256, 2)
__global__ void gemm_bf16_kernel(const __bf16* __restrict__ A, const __bf16* __restrict__ Bt,
                                 int M, int N, int K,
                                 const float* __restrict__ bias,
                                 const float* __restrict__ resid,
                                 int doGelu,
                                 float* __restrict__ outF,
                                 __bf16* __restrict__ outB) {
    __shared__ __bf16 As[2][BM][BK + PAD];
    __shared__ __bf16 Bs[2][BN][BK + PAD];

    const int tid  = threadIdx.x;
    const int lane = tid & 31;
    const int wid  = tid >> 5;      // 8 waves
    const int wm   = wid & 3;       // 4 along M (32 rows each)
    const int wn   = wid >> 2;      // 2 along N (64 cols each)
    const int col  = lane & 15;
    const int half = lane >> 4;

    const int bm = blockIdx.x * BM;
    const int bn = blockIdx.y * BN;
    const int nk = K / BK;

    // each thread stages 2 A-chunks + 2 B-chunks of 16B per tile (4 async ops/lane)
    auto load_tile = [&](int buf, int k0) {
#pragma unroll
        for (int i = 0; i < 2; ++i) {
            const int ch = tid * 2 + i;       // 0..511
            const int r  = ch >> 2;           // 0..127
            const int kc = (ch & 3) * 8;      // 0,8,16,24
            copy16(A  + (size_t)(bm + r) * K + k0 + kc, &As[buf][r][kc]);
            copy16(Bt + (size_t)(bn + r) * K + k0 + kc, &Bs[buf][r][kc]);
        }
    };

    v8f zero = {};
    v8f acc[2][4];
#pragma unroll
    for (int mi = 0; mi < 2; ++mi)
#pragma unroll
        for (int ni = 0; ni < 4; ++ni) acc[mi][ni] = zero;

    load_tile(0, 0);

    for (int kt = 0; kt < nk; ++kt) {
        const int buf = kt & 1;
        if (kt + 1 < nk) {
            load_tile(buf ^ 1, (kt + 1) * BK);
#if USE_ASYNC_LDS
            __builtin_amdgcn_s_wait_asynccnt(4);  // previous tile's 4 ops retired
#endif
        } else {
#if USE_ASYNC_LDS
            __builtin_amdgcn_s_wait_asynccnt(0);
#endif
        }
        __syncthreads();

        Frag af[2], bf[4];
#pragma unroll
        for (int mi = 0; mi < 2; ++mi) {
            const int r = wm * 32 + mi * 16 + col;
#pragma unroll
            for (int p = 0; p < 8; ++p)
                af[mi].u[p] = *(const u32*)&As[buf][r][afrag_base(p, half)];
        }
#pragma unroll
        for (int ni = 0; ni < 4; ++ni) {
            const int n = wn * 64 + ni * 16 + col;
#pragma unroll
            for (int p = 0; p < 8; ++p)
                bf[ni].u[p] = *(const u32*)&Bs[buf][n][half * 16 + 2 * p];
        }
#pragma unroll
        for (int mi = 0; mi < 2; ++mi)
#pragma unroll
            for (int ni = 0; ni < 4; ++ni)
                acc[mi][ni] = __builtin_amdgcn_wmma_f32_16x16x32_bf16(
                    false, af[mi].v, false, bf[ni].v, (short)0, acc[mi][ni], false, false);
        __syncthreads();   // all waves done reading buf before it is overwritten
    }

    // epilogue: C/D layout — VGPR i holds row (i + half*8), col = lane&15
#pragma unroll
    for (int mi = 0; mi < 2; ++mi)
#pragma unroll
        for (int ni = 0; ni < 4; ++ni)
#pragma unroll
            for (int i = 0; i < 8; ++i) {
                const int gr = bm + wm * 32 + mi * 16 + half * 8 + i;
                const int gc = bn + wn * 64 + ni * 16 + col;
                float v = acc[mi][ni][i];
                if (bias)   v += bias[gc];
                if (doGelu) v = gelu_tanh_f(v);
                if (resid)  v += resid[(size_t)gr * N + gc];
                if (outF)   outF[(size_t)gr * N + gc] = v;
                if (outB)   outB[(size_t)gr * N + gc] = (__bf16)v;
            }
}

// ---------------- flash attention (causal), bf16 QKV, bf16 out ----------------
#define QT 64

__launch_bounds__(128, 2)
__global__ void attn_kernel(const __bf16* __restrict__ Q, const __bf16* __restrict__ Kt,
                            const __bf16* __restrict__ V, __bf16* __restrict__ O) {
    __shared__ __bf16 Ks[QT][HDm + PAD];   // Ks[key][d] : async-load target, B-frag reads d
    __shared__ __bf16 Vs[HDm][QT + 2];     // Vs[d][key] : manual transpose
    __shared__ __bf16 Ps[4][16][QT];       // per-wave P pane (C-layout -> A-layout relay)

    const int tid = threadIdx.x, lane = tid & 31, wid = tid >> 5;
    const int col = lane & 15, half = lane >> 4;
    const int qt = blockIdx.x;
    const int bh = blockIdx.y;
    const int b  = bh >> 4, h = bh & 15;
    const size_t base = (size_t)b * Sm * Dm + (size_t)h * HDm;

    // Q fragments for this wave's 16 rows (K-dim = HD = 64 -> 2 steps)
    Frag aq[2];
    {
        const int qr = qt * QT + wid * 16 + col;
#pragma unroll
        for (int ks = 0; ks < 2; ++ks)
#pragma unroll
            for (int p = 0; p < 8; ++p)
                aq[ks].u[p] = *(const u32*)&Q[base + (size_t)qr * Dm + ks * 32 + afrag_base(p, half)];
    }

    v8f zero = {};
    v8f o[4]; o[0] = zero; o[1] = zero; o[2] = zero; o[3] = zero;
    float mrow[8], lrow[8];
#pragma unroll
    for (int i = 0; i < 8; ++i) { mrow[i] = -3.0e38f; lrow[i] = 0.f; }

    const int key = tid >> 1;
    const int ds  = (tid & 1) * 32;

    for (int kt = 0; kt <= qt; ++kt) {
        __syncthreads();
        // K tile 64x64 — natural layout, async global->LDS
#pragma unroll
        for (int i = 0; i < 4; ++i)
            copy16(Kt + base + (size_t)(kt * QT + key) * Dm + ds + i * 8, &Ks[key][ds + i * 8]);
        // V tile 64x64 — transposed into LDS (manual)
        {
            __bf16 tmp[32];
            const u32* s32 = (const u32*)(V + base + (size_t)(kt * QT + key) * Dm + ds);
            u32* t32 = (u32*)tmp;
#pragma unroll
            for (int i = 0; i < 16; ++i) t32[i] = s32[i];
#pragma unroll
            for (int j = 0; j < 32; ++j) Vs[ds + j][key] = tmp[j];
        }
#if USE_ASYNC_LDS
        __builtin_amdgcn_s_wait_asynccnt(0);
#endif
        __syncthreads();

        // scores: S(16x64) = Q(16x64) @ K^T, 4 N-tiles x 2 chained K=32 WMMAs
        v8f st[4];
#pragma unroll
        for (int ni = 0; ni < 4; ++ni) {
            Frag bk0, bk1;
            const int n = ni * 16 + col;
#pragma unroll
            for (int p = 0; p < 8; ++p) {
                bk0.u[p] = *(const u32*)&Ks[n][half * 16 + 2 * p];
                bk1.u[p] = *(const u32*)&Ks[n][32 + half * 16 + 2 * p];
            }
            v8f c = zero;
            c = __builtin_amdgcn_wmma_f32_16x16x32_bf16(false, aq[0].v, false, bk0.v, (short)0, c, false, false);
            c = __builtin_amdgcn_wmma_f32_16x16x32_bf16(false, aq[1].v, false, bk1.v, (short)0, c, false, false);
            st[ni] = c * 0.125f;   // 1/sqrt(64)
        }
        // causal mask
#pragma unroll
        for (int ni = 0; ni < 4; ++ni)
#pragma unroll
            for (int i = 0; i < 8; ++i) {
                const int kidx = kt * QT + ni * 16 + col;
                const int qidx = qt * QT + wid * 16 + half * 8 + i;
                if (kidx > qidx) st[ni][i] = -3.0e38f;
            }
        // online softmax (row lives across the 16 lanes of one half)
#pragma unroll
        for (int i = 0; i < 8; ++i) {
            float mx = fmaxf(fmaxf(st[0][i], st[1][i]), fmaxf(st[2][i], st[3][i]));
#pragma unroll
            for (int d = 1; d < 16; d <<= 1) mx = fmaxf(mx, __shfl_xor(mx, d, 32));
            const float mnew = fmaxf(mrow[i], mx);
            const float corr = __expf(mrow[i] - mnew);
            float rs = 0.f;
#pragma unroll
            for (int ni = 0; ni < 4; ++ni) {
                const float p = __expf(st[ni][i] - mnew);
                st[ni][i] = p; rs += p;
            }
#pragma unroll
            for (int d = 1; d < 16; d <<= 1) rs += __shfl_xor(rs, d, 32);
            lrow[i] = lrow[i] * corr + rs;
            mrow[i] = mnew;
#pragma unroll
            for (int nd = 0; nd < 4; ++nd) o[nd][i] *= corr;
        }
        // relay P through per-wave LDS pane: C-layout -> A-fragment layout
#pragma unroll
        for (int ni = 0; ni < 4; ++ni)
#pragma unroll
            for (int i = 0; i < 8; ++i)
                Ps[wid][half * 8 + i][ni * 16 + col] = (__bf16)st[ni][i];

        Frag pa[2];
#pragma unroll
        for (int ks = 0; ks < 2; ++ks)
#pragma unroll
            for (int p = 0; p < 8; ++p)
                pa[ks].u[p] = *(const u32*)&Ps[wid][col][ks * 32 + afrag_base(p, half)];

        // O += P(16x64) @ V(64x64)
#pragma unroll
        for (int nd = 0; nd < 4; ++nd) {
            Frag bv0, bv1;
            const int dcol = nd * 16 + col;
#pragma unroll
            for (int p = 0; p < 8; ++p) {
                bv0.u[p] = *(const u32*)&Vs[dcol][half * 16 + 2 * p];
                bv1.u[p] = *(const u32*)&Vs[dcol][32 + half * 16 + 2 * p];
            }
            o[nd] = __builtin_amdgcn_wmma_f32_16x16x32_bf16(false, pa[0].v, false, bv0.v, (short)0, o[nd], false, false);
            o[nd] = __builtin_amdgcn_wmma_f32_16x16x32_bf16(false, pa[1].v, false, bv1.v, (short)0, o[nd], false, false);
        }
    }

#pragma unroll
    for (int nd = 0; nd < 4; ++nd)
#pragma unroll
        for (int i = 0; i < 8; ++i) {
            const float val = o[nd][i] / lrow[i];
            O[base + (size_t)(qt * QT + wid * 16 + half * 8 + i) * Dm + nd * 16 + col] = (__bf16)val;
        }
}

// ---------------- host-side launcher ----------------
extern "C" void kernel_launch(void* const* d_in, const int* in_sizes, int n_in,
                              void* d_out, int out_size, void* d_ws, size_t ws_size,
                              hipStream_t stream) {
    (void)in_sizes; (void)n_in; (void)out_size; (void)ws_size;
    const float* x   = (const float*)d_in[0];
    const float* Wq  = (const float*)d_in[1];
    const float* Wk  = (const float*)d_in[2];
    const float* Wv  = (const float*)d_in[3];
    const float* Wo  = (const float*)d_in[4];
    const float* bo  = (const float*)d_in[5];
    const float* W1  = (const float*)d_in[6];
    const float* b1  = (const float*)d_in[7];
    const float* W2  = (const float*)d_in[8];
    const float* b2  = (const float*)d_in[9];
    const float* g1  = (const float*)d_in[10];
    const float* be1 = (const float*)d_in[11];
    const float* g2  = (const float*)d_in[12];
    const float* be2 = (const float*)d_in[13];
    float* out = (float*)d_out;

    char* wp = (char*)d_ws;
    auto carve = [&](size_t bytes) {
        char* p = wp;
        wp += (bytes + 255) & ~(size_t)255;
        return (void*)p;
    };
    // persistent region: transposed bf16 weights [N][K] + fp32 x1
    __bf16* WqT = (__bf16*)carve((size_t)Dm * Dm * 2);
    __bf16* WkT = (__bf16*)carve((size_t)Dm * Dm * 2);
    __bf16* WvT = (__bf16*)carve((size_t)Dm * Dm * 2);
    __bf16* WoT = (__bf16*)carve((size_t)Dm * Dm * 2);
    __bf16* W1T = (__bf16*)carve((size_t)Dm * DFF * 2);
    __bf16* W2T = (__bf16*)carve((size_t)DFF * Dm * 2);
    float*  x1  = (float*)carve((size_t)ROWS * Dm * 4);
    // overlay region A (attention phase)
    char* ovl = wp;
    __bf16* h1 = (__bf16*)carve((size_t)ROWS * Dm * 2);
    __bf16* Qb = (__bf16*)carve((size_t)ROWS * Dm * 2);
    __bf16* Kb = (__bf16*)carve((size_t)ROWS * Dm * 2);
    __bf16* Vb = (__bf16*)carve((size_t)ROWS * Dm * 2);
    __bf16* Ab = (__bf16*)carve((size_t)ROWS * Dm * 2);
    // overlay region B (FFN phase) aliases region A — safe: sequential stream order
    wp = ovl;
    __bf16* h2 = (__bf16*)carve((size_t)ROWS * Dm * 2);
    __bf16* u  = (__bf16*)carve((size_t)ROWS * DFF * 2);

    // 1) weights fp32 -> bf16, transposed to [N][K] so GEMM B-tiles are K-contiguous
    dim3 tb(32, 8);
    cast_transpose_bf16_kernel<<<dim3(Dm / 32, Dm / 32), tb, 0, stream>>>(Wq, WqT, Dm, Dm);
    cast_transpose_bf16_kernel<<<dim3(Dm / 32, Dm / 32), tb, 0, stream>>>(Wk, WkT, Dm, Dm);
    cast_transpose_bf16_kernel<<<dim3(Dm / 32, Dm / 32), tb, 0, stream>>>(Wv, WvT, Dm, Dm);
    cast_transpose_bf16_kernel<<<dim3(Dm / 32, Dm / 32), tb, 0, stream>>>(Wo, WoT, Dm, Dm);
    cast_transpose_bf16_kernel<<<dim3(DFF / 32, Dm / 32), tb, 0, stream>>>(W1, W1T, Dm, DFF);
    cast_transpose_bf16_kernel<<<dim3(Dm / 32, DFF / 32), tb, 0, stream>>>(W2, W2T, DFF, Dm);

    // 2) h1 = LN(x)
    ln_bf16_kernel<<<ROWS, 256, 0, stream>>>(x, g1, be1, h1);

    // 3) Q/K/V projections (bf16 out)
    dim3 gQKV(ROWS / BM, Dm / BN);
    gemm_bf16_kernel<<<gQKV, 256, 0, stream>>>(h1, WqT, ROWS, Dm, Dm, nullptr, nullptr, 0, nullptr, Qb);
    gemm_bf16_kernel<<<gQKV, 256, 0, stream>>>(h1, WkT, ROWS, Dm, Dm, nullptr, nullptr, 0, nullptr, Kb);
    gemm_bf16_kernel<<<gQKV, 256, 0, stream>>>(h1, WvT, ROWS, Dm, Dm, nullptr, nullptr, 0, nullptr, Vb);

    // 4) causal flash attention
    dim3 gAtt(Sm / QT, Bb * Hn);
    attn_kernel<<<gAtt, 128, 0, stream>>>(Qb, Kb, Vb, Ab);

    // 5) x1 = x + attn @ Wo + bo   (fp32)
    gemm_bf16_kernel<<<gQKV, 256, 0, stream>>>(Ab, WoT, ROWS, Dm, Dm, bo, x, 0, x1, nullptr);

    // 6) h2 = LN(x1)
    ln_bf16_kernel<<<ROWS, 256, 0, stream>>>(x1, g2, be2, h2);

    // 7) u = gelu(h2 @ W1 + b1)    (bf16)
    dim3 gF1(ROWS / BM, DFF / BN);
    gemm_bf16_kernel<<<gF1, 256, 0, stream>>>(h2, W1T, ROWS, DFF, Dm, b1, nullptr, 1, nullptr, u);

    // 8) out = x1 + u @ W2 + b2    (fp32)
    dim3 gF2(ROWS / BM, Dm / BN);
    gemm_bf16_kernel<<<gF2, 256, 0, stream>>>(u, W2T, ROWS, Dm, DFF, b2, x1, 0, out, nullptr);
}
